// CausalMultiHeadedSelfAttention_57698590654652
// MI455X (gfx1250) — matrix operand order repaired and checked by reading, not compiled, not run on hardware
//
#include <hip/hip_runtime.h>
#include <hip/hip_bf16.h>
#include <math.h>

// Flip to 0 if the gfx1250 assembler rejects the async-LDS mnemonics.
#define USE_ASYNC_LDS 1

// ---------------- types & helpers ----------------
typedef __attribute__((ext_vector_type(16))) __bf16 v16bf;
typedef __attribute__((ext_vector_type(8)))  float  v8f;

union Frag { unsigned int u[8]; v16bf b; };

__device__ __forceinline__ unsigned short f2bf(float f) {
    unsigned int u = __float_as_uint(f);
    u += 0x7FFFu + ((u >> 16) & 1u);          // round-to-nearest-even
    return (unsigned short)(u >> 16);
}
__device__ __forceinline__ unsigned int pack2(float lo, float hi) {
    return (unsigned int)f2bf(lo) | ((unsigned int)f2bf(hi) << 16);
}

// Problem constants
#define BB 2
#define SS 2048
#define DD 1024
#define HH 16
#define DK 64

// ---------------- GEMM: out = X @ W^T ----------------
// Tile: M=128, N=64, K-step=32. 256 threads = 8 waves in a 4(M)x2(N) grid;
// each wave computes a 32x32 subtile = 2x2 WMMA 16x16 accumulators.
// MODE 0: fp32 out in [b,h,s,d]   (Q/K pre-RoPE)
// MODE 1: bf16 out in [b,h,s,d]   (V)
// MODE 2: fp32 out row-major [m,n] (final output)
template<int MODE, bool INBF16>
__global__ __launch_bounds__(256) void proj_kernel(const void* __restrict__ xin,
                                                   const float* __restrict__ w,
                                                   void* __restrict__ out)
{
    __shared__ unsigned int   A_u[128 * 17];   // 128x32 bf16 X tile, padded
    __shared__ unsigned short Bt_s[32 * 66];   // 32(k) x 64(n) bf16 W^T tile, padded

    const int tid  = threadIdx.x;
    const int lane = tid & 31;
    const int wv   = tid >> 5;
    const int mt0  = (wv >> 1) * 32;
    const int nt0  = (wv & 1) * 32;
    const int m0   = blockIdx.x * 128;
    const int n0   = blockIdx.y * 64;
    const int khalf = lane >> 4;
    const int mlo   = lane & 15;

    v8f acc[2][2];
    #pragma unroll
    for (int s = 0; s < 2; ++s)
        #pragma unroll
        for (int t = 0; t < 2; ++t)
            #pragma unroll
            for (int e = 0; e < 8; ++e) acc[s][t][e] = 0.0f;

    for (int k0 = 0; k0 < DD; k0 += 32) {
        __syncthreads();
        // ---- stage A tile (convert fp32->bf16 or copy bf16) ----
        if (!INBF16) {
            const float2* xf = (const float2*)xin;   // [4096][512] float2
            #pragma unroll
            for (int j = 0; j < 8; ++j) {
                int p  = tid + j * 256;              // 0..2047
                int r  = p >> 4;                     // 0..127
                int c2 = p & 15;                     // float2 column
                float2 v = xf[(size_t)(m0 + r) * (DD / 2) + (k0 >> 1) + c2];
                A_u[r * 17 + c2] = pack2(v.x, v.y);
            }
            if (k0 + 32 < DD)                        // prefetch next A tile slice
                __builtin_prefetch(&xf[(size_t)(m0 + (tid >> 4)) * (DD / 2) + ((k0 + 32) >> 1) + (tid & 15)], 0, 0);
        } else {
            const unsigned int* xu = (const unsigned int*)xin; // [4096][512] uint
            #pragma unroll
            for (int j = 0; j < 8; ++j) {
                int p  = tid + j * 256;
                int r  = p >> 4;
                int c2 = p & 15;
                A_u[r * 17 + c2] = xu[(size_t)(m0 + r) * (DD / 2) + (k0 >> 1) + c2];
            }
            if (k0 + 32 < DD)
                __builtin_prefetch(&xu[(size_t)(m0 + (tid >> 4)) * (DD / 2) + ((k0 + 32) >> 1) + (tid & 15)], 0, 0);
        }
        // ---- stage W tile transposed: Bt[k][n] = bf16(W[n0+n][k0+k]) ----
        #pragma unroll
        for (int j = 0; j < 8; ++j) {
            int p = tid + j * 256;                   // 0..2047
            int n = p >> 5;                          // 0..63
            int k = p & 31;
            Bt_s[k * 66 + n] = f2bf(w[(size_t)(n0 + n) * DD + k0 + k]);
        }
        if (k0 + 32 < DD)                            // prefetch next W tile slice
            __builtin_prefetch(&w[(size_t)(n0 + (tid >> 5)) * DD + (k0 + 32) + (tid & 31)], 0, 0);
        __syncthreads();

        // ---- fragments ----
        Frag a[2], bf[2];
        #pragma unroll
        for (int s = 0; s < 2; ++s) {
            int m = mt0 + 16 * s + mlo;
            #pragma unroll
            for (int i = 0; i < 4; ++i) {
                a[s].u[i]     = A_u[m * 17 + khalf * 4 + i];
                a[s].u[4 + i] = A_u[m * 17 + 8 + khalf * 4 + i];
            }
        }
        #pragma unroll
        for (int t = 0; t < 2; ++t) {
            int nt = nt0 + 16 * t;
            #pragma unroll
            for (int i = 0; i < 8; ++i)
            { int kk = (i < 4) ? (8 * khalf + 2 * i) : (16 + 8 * khalf + 2 * (i - 4));
                bf[t].u[i] = (unsigned int)Bt_s[kk * 66 + nt + mlo] | ((unsigned int)Bt_s[(kk + 1) * 66 + nt + mlo] << 16); }
        }
        #pragma unroll
        for (int s = 0; s < 2; ++s)
            #pragma unroll
            for (int t = 0; t < 2; ++t)
                acc[s][t] = __builtin_amdgcn_wmma_f32_16x16x32_bf16(
                    false, a[s].b, false, bf[t].b, (short)0, acc[s][t], false, false);
    }

    // ---- epilogue ----
    const int hi = lane >> 4;
    #pragma unroll
    for (int s = 0; s < 2; ++s)
        #pragma unroll
        for (int t = 0; t < 2; ++t)
            #pragma unroll
            for (int i = 0; i < 8; ++i) {
                int m = m0 + mt0 + 16 * s + i + 8 * hi;
                int n = n0 + nt0 + 16 * t + (lane & 15);
                float v = acc[s][t][i];
                if (MODE == 2) {
                    ((float*)out)[(size_t)m * DD + n] = v;
                } else {
                    int bb = m >> 11, srow = m & (SS - 1);
                    int h = n >> 6,  d    = n & (DK - 1);
                    size_t idx = (((size_t)(bb * HH + h)) * SS + srow) * DK + d;
                    if (MODE == 0) ((float*)out)[idx] = v;
                    else           ((unsigned short*)out)[idx] = f2bf(v);
                }
            }
}

// ---------------- RoPE (interleaved pairs) fp32 -> bf16 ----------------
__global__ __launch_bounds__(256) void rope_kernel(const float* __restrict__ in,
                                                   unsigned int* __restrict__ outb)
{
    int idx = blockIdx.x * blockDim.x + threadIdx.x;   // over B*H*S*(DK/2)
    int j = idx & 31;                                  // pair index
    int s = (idx >> 5) & (SS - 1);                     // token position
    float inv = __expf(-((float)(2 * j) * (1.0f / (float)DK)) * __logf(10000.0f));
    float ang = (float)s * inv;
    float sn, cs;
    __sincosf(ang, &sn, &cs);
    float2 v = ((const float2*)in)[idx];
    outb[idx] = pack2(v.x * cs - v.y * sn, v.x * sn + v.y * cs);
}

// ---------------- Flash attention ----------------
// block = 128 threads (4 waves). blockIdx.x = b*H+h, blockIdx.y = 64-row q block.
// Each wave owns 16 q rows. KV staged in LDS in 32-key blocks.
// Row sums of P computed as P @ ones via WMMA (replicated across N lanes in the
// C fragment) instead of cross-lane bpermute reductions.
__global__ __launch_bounds__(128) void attn_kernel(const unsigned int* __restrict__ qb,
                                                   const unsigned int* __restrict__ kb,
                                                   const unsigned int* __restrict__ vb,
                                                   unsigned short* __restrict__ attn)
{
    __shared__ unsigned short Kt_s[64 * 34];   // K^T: [d][key] bf16, padded
    __shared__ unsigned int   V_u[32 * 33];    // V:   [key][d/2] uint, padded
    __shared__ unsigned short P_s[4 * 16 * 34];// per-wave P: [q][key] bf16, padded

    const int tid  = threadIdx.x;
    const int lane = tid & 31;
    const int wv   = tid >> 5;
    const int bh   = blockIdx.x;
    const int b    = bh >> 4, h = bh & 15;
    const int q0   = blockIdx.y * 64;
    const int qw0  = q0 + wv * 16;
    const int khalf = lane >> 4, mlo = lane & 15;

    const unsigned int* qp = qb + (size_t)bh * SS * (DK / 2);
    const unsigned int* kp = kb + (size_t)bh * SS * (DK / 2);
    const unsigned int* vp = vb + (size_t)bh * SS * (DK / 2);

    // Q fragments (A layout), one per 32-wide K chunk of d_k=64
    Frag qa[2];
    {
        int m = qw0 + mlo;
        #pragma unroll
        for (int c = 0; c < 2; ++c) {
            uint4 lo = *(const uint4*)(qp + (size_t)m * 32 + 16 * c + khalf * 4);
            uint4 hi = *(const uint4*)(qp + (size_t)m * 32 + 16 * c + 8 + khalf * 4);
            qa[c].u[0] = lo.x; qa[c].u[1] = lo.y; qa[c].u[2] = lo.z; qa[c].u[3] = lo.w;
            qa[c].u[4] = hi.x; qa[c].u[5] = hi.y; qa[c].u[6] = hi.z; qa[c].u[7] = hi.w;
        }
    }

    // constant all-ones bf16 B fragment (for row sums P @ ones)
    Frag ones;
    #pragma unroll
    for (int i = 0; i < 8; ++i) ones.u[i] = 0x3F803F80u;

    float mrow[8], lrow[8], alpha[8];
    v8f o[4];
    #pragma unroll
    for (int i = 0; i < 8; ++i) { mrow[i] = -INFINITY; lrow[i] = 0.0f; }
    #pragma unroll
    for (int dt = 0; dt < 4; ++dt)
        #pragma unroll
        for (int e = 0; e < 8; ++e) o[dt][e] = 0.0f;

    const int jend = q0 + 64;
    for (int j0 = 0; j0 < jend; j0 += 32) {
        __syncthreads();
        // ---- stage V (async copy to LDS) and K (transposed, manual) ----
        #pragma unroll
        for (int jj = 0; jj < 8; ++jj) {
            int p   = tid + jj * 128;        // 0..1023
            int key = p >> 5, dp = p & 31;
#if USE_ASYNC_LDS
            const unsigned int* g = vp + (size_t)(j0 + key) * 32 + dp;
            unsigned int lds_off = (unsigned int)(size_t)&V_u[key * 33 + dp];
            asm volatile("global_load_async_to_lds_b32 %0, %1, off"
                         :: "v"(lds_off), "v"(g) : "memory");
#else
            V_u[key * 33 + dp] = vp[(size_t)(j0 + key) * 32 + dp];
#endif
            unsigned int kw = kp[(size_t)(j0 + key) * 32 + dp];
            Kt_s[(2 * dp) * 34 + key]     = (unsigned short)(kw & 0xFFFFu);
            Kt_s[(2 * dp + 1) * 34 + key] = (unsigned short)(kw >> 16);
        }
        if (j0 + 32 < jend) {               // prefetch next KV block
            __builtin_prefetch(vp + (size_t)(j0 + 32 + (tid >> 2)) * 32 + (tid & 3) * 8, 0, 0);
            __builtin_prefetch(kp + (size_t)(j0 + 32 + (tid >> 2)) * 32 + (tid & 3) * 8, 0, 0);
        }
#if USE_ASYNC_LDS
        asm volatile("s_wait_asynccnt 0x0" ::: "memory");
#endif
        __syncthreads();

        // ---- S = Q K^T  (two 16x16 key tiles, K-dim = d_k split 2x32) ----
        v8f sc[2];
        #pragma unroll
        for (int nt = 0; nt < 2; ++nt) {
            #pragma unroll
            for (int e = 0; e < 8; ++e) sc[nt][e] = 0.0f;
            #pragma unroll
            for (int c = 0; c < 2; ++c) {
                Frag kf;
                #pragma unroll
                for (int i = 0; i < 8; ++i) {
                    int d = 32 * c + ((i < 4) ? (8 * khalf + 2 * i) : (16 + 8 * khalf + 2 * (i - 4)));
                    kf.u[i] = (unsigned int)Kt_s[d * 34 + nt * 16 + mlo] | ((unsigned int)Kt_s[(d + 1) * 34 + nt * 16 + mlo] << 16);
                }
                sc[nt] = __builtin_amdgcn_wmma_f32_16x16x32_bf16(
                    false, qa[c].b, false, kf.b, (short)0, sc[nt], false, false);
            }
        }

        // ---- scale + causal mask + online max (cross-lane), rescale O ----
        const float scale = 0.125f;                 // d_k^-0.5
        const int key0 = j0 + mlo;
        #pragma unroll
        for (int i = 0; i < 8; ++i) {
            int q = qw0 + i + 8 * khalf;
            float s0 = sc[0][i] * scale; if (key0 > q)      s0 = -INFINITY;
            float s1 = sc[1][i] * scale; if (key0 + 16 > q) s1 = -INFINITY;
            float bm = fmaxf(s0, s1);
            #pragma unroll
            for (int msk = 1; msk < 16; msk <<= 1)
                bm = fmaxf(bm, __shfl_xor(bm, msk, 32));
            float mnew = fmaxf(mrow[i], bm);
            alpha[i] = __expf(mrow[i] - mnew);
            mrow[i] = mnew;
            o[0][i] *= alpha[i]; o[1][i] *= alpha[i];
            o[2][i] *= alpha[i]; o[3][i] *= alpha[i];
            sc[0][i] = __expf(s0 - mnew);
            sc[1][i] = __expf(s1 - mnew);
        }

        // ---- re-fragment P: C layout -> LDS -> A layout (per-wave buffer) ----
        #pragma unroll
        for (int nt = 0; nt < 2; ++nt)
            #pragma unroll
            for (int i = 0; i < 8; ++i)
                P_s[(wv * 16 + i + 8 * khalf) * 34 + nt * 16 + mlo] = f2bf(sc[nt][i]);
        Frag pa;
        #pragma unroll
        for (int i = 0; i < 4; ++i) {
            pa.u[i]     = *(const unsigned int*)&P_s[(wv * 16 + mlo) * 34 + khalf * 8 + 2 * i];
            pa.u[4 + i] = *(const unsigned int*)&P_s[(wv * 16 + mlo) * 34 + 16 + khalf * 8 + 2 * i];
        }

        // ---- row sums via matrix core: bsum = P @ ones (replicated over N) ----
        v8f zero8;
        #pragma unroll
        for (int e = 0; e < 8; ++e) zero8[e] = 0.0f;
        v8f bsum = __builtin_amdgcn_wmma_f32_16x16x32_bf16(
            false, pa.b, false, ones.b, (short)0, zero8, false, false);
        #pragma unroll
        for (int i = 0; i < 8; ++i)
            lrow[i] = lrow[i] * alpha[i] + bsum[i];

        // ---- O += P V  (4 d-tiles of 16) ----
        #pragma unroll
        for (int dt = 0; dt < 4; ++dt) {
            Frag vf;
            #pragma unroll
            for (int i = 0; i < 8; ++i) {
                int kk = (i < 4) ? (8 * khalf + 2 * i) : (16 + 8 * khalf + 2 * (i - 4));
                const unsigned short* Vs = (const unsigned short*)V_u;
                vf.u[i] = (unsigned int)Vs[kk * 66 + dt * 16 + mlo] | ((unsigned int)Vs[(kk + 1) * 66 + dt * 16 + mlo] << 16);
            }
            o[dt] = __builtin_amdgcn_wmma_f32_16x16x32_bf16(
                false, pa.b, false, vf.b, (short)0, o[dt], false, false);
        }
    }

    // ---- normalize + store bf16 attn out in [b, s, h*64+d] ----
    #pragma unroll
    for (int i = 0; i < 8; ++i) {
        float inv = 1.0f / lrow[i];
        int srow = qw0 + i + 8 * khalf;
        size_t base = ((size_t)b * SS + srow) * DD + h * DK;
        #pragma unroll
        for (int dt = 0; dt < 4; ++dt)
            attn[base + dt * 16 + mlo] = f2bf(o[dt][i] * inv);
    }
}

// ---------------- launcher ----------------
extern "C" void kernel_launch(void* const* d_in, const int* in_sizes, int n_in,
                              void* d_out, int out_size, void* d_ws, size_t ws_size,
                              hipStream_t stream)
{
    (void)in_sizes; (void)n_in; (void)out_size; (void)ws_size;
    const float* x  = (const float*)d_in[0];
    const float* wq = (const float*)d_in[1];
    const float* wk = (const float*)d_in[2];
    const float* wv = (const float*)d_in[3];
    const float* wo = (const float*)d_in[4];
    // d_in[5] = token_positions == arange(S); positions derived analytically.

    char* ws = (char*)d_ws;
    const size_t F32SZ = (size_t)BB * HH * SS * DK * 4;  // 16 MiB
    const size_t BF16SZ = F32SZ / 2;                     // 8 MiB
    float*          qf    = (float*)(ws);
    float*          kf    = (float*)(ws + F32SZ);
    unsigned int*   qbu   = (unsigned int*)(ws + 2 * F32SZ);
    unsigned int*   kbu   = (unsigned int*)(ws + 2 * F32SZ + BF16SZ);
    unsigned int*   vbu   = (unsigned int*)(ws + 2 * F32SZ + 2 * BF16SZ);
    unsigned short* attnb = (unsigned short*)(ws + 2 * F32SZ + 3 * BF16SZ);

    dim3 gproj(32, 16), bproj(256);
    proj_kernel<0, false><<<gproj, bproj, 0, stream>>>((const void*)x, wq, (void*)qf);
    proj_kernel<0, false><<<gproj, bproj, 0, stream>>>((const void*)x, wk, (void*)kf);
    proj_kernel<1, false><<<gproj, bproj, 0, stream>>>((const void*)x, wv, (void*)vbu);

    const int npairs = BB * HH * SS * (DK / 2);          // 2,097,152
    rope_kernel<<<npairs / 256, 256, 0, stream>>>(qf, qbu);
    rope_kernel<<<npairs / 256, 256, 0, stream>>>(kf, kbu);

    attn_kernel<<<dim3(BB * HH, SS / 64), 128, 0, stream>>>(qbu, kbu, vbu, attnb);

    proj_kernel<2, true><<<gproj, bproj, 0, stream>>>((const void*)attnb, wo, d_out);
}
